// MixtralDecoderLayer_39238821216264
// MI455X (gfx1250) — matrix operand order repaired and checked by reading, not compiled, not run on hardware
//
#include <hip/hip_runtime.h>

// ---------------- problem constants ----------------
#define Bc   2
#define Sc   2048
#define Dc   2048
#define NHc  16
#define NKVc 8
#define HDc  128
#define Ec   8
#define Kc   2
#define Fc   4096
#define Tc   (Bc * Sc)                 // 4096 tokens
#define QKVF ((NHc + 2 * NKVc) * HDc)  // 4096

typedef __attribute__((ext_vector_type(16))) __bf16 v16bf;
typedef __attribute__((ext_vector_type(8)))  float  v8f;
typedef __attribute__((ext_vector_type(4)))  int    v4i;

#define LDS_STRIDE 40  // 32 + 8 pad, 80B row stride (16B aligned)

__device__ __forceinline__ v8f v8f_zero() {
  v8f z;
  for (int i = 0; i < 8; ++i) z[i] = 0.0f;
  return z;
}

__device__ __forceinline__ v8f wmma_bf16(v16bf a, v16bf b, v8f c) {
  return __builtin_amdgcn_wmma_f32_16x16x32_bf16(
      /*neg_a=*/false, a, /*neg_b=*/false, b,
      /*c_mod=*/(short)0, c, /*reuse_a=*/false, /*reuse_b=*/false);
}

// ---------------- CDNA5 async global->LDS copy (ASYNCcnt path) ----------------
#if defined(__has_builtin)
#if __has_builtin(__builtin_amdgcn_global_load_async_to_lds_b128)
#define HAVE_ASYNC_BUILTIN 1
#endif
#if __has_builtin(__builtin_amdgcn_s_wait_asynccnt)
#define HAVE_ASYNCWAIT_BUILTIN 1
#endif
#endif

__device__ __forceinline__ void async_cp16(const __bf16* g, __bf16* l) {
#if defined(HAVE_ASYNC_BUILTIN)
  __builtin_amdgcn_global_load_async_to_lds_b128(
      (__attribute__((address_space(1))) v4i*)g,
      (__attribute__((address_space(3))) v4i*)l, 0, 0);
#else
  // GV mode: vdst = LDS byte address, vaddr = 64-bit global address
  asm volatile("global_load_async_to_lds_b128 %0, %1, off"
               :: "v"((unsigned)(uintptr_t)l), "v"(g)
               : "memory");
#endif
}

__device__ __forceinline__ void async_wait0() {
#if defined(HAVE_ASYNCWAIT_BUILTIN)
  __builtin_amdgcn_s_wait_asynccnt(0);
#else
  asm volatile("s_wait_asynccnt 0x0" ::: "memory");
#endif
}

// A fragment (16x32 bf16): lane m = l&15, k chunks at (l>>4)*8 and +16
__device__ __forceinline__ v16bf frag_a(const __bf16* lds, int mbase, int lane) {
  int m  = mbase + (lane & 15);
  int k0 = (lane >> 4) * 8;
  v16bf a;
  ((uint4*)&a)[0] = *(const uint4*)(lds + m * LDS_STRIDE + k0);
  ((uint4*)&a)[1] = *(const uint4*)(lds + m * LDS_STRIDE + k0 + 16);
  return a;
}

// B fragment (32x16 bf16) from [n][k] LDS tile: lane n = l&15, k chunks at (l>>4)*8 and +16
__device__ __forceinline__ v16bf frag_b(const __bf16* lds, int nbase, int lane) {
  int n  = nbase + (lane & 15);
  int k0 = (lane >> 4) * 8;
  v16bf b;
  ((uint4*)&b)[0] = *(const uint4*)(lds + n * LDS_STRIDE + k0);
  ((uint4*)&b)[1] = *(const uint4*)(lds + n * LDS_STRIDE + k0 + 16);
  return b;
}

// async-stage a 128x32 bf16 tile (row-major rows, leading dim ld) into LDS
__device__ __forceinline__ void stage_rows_async(const __bf16* src_base, size_t ld, int k0,
                                                 __bf16* lds, int tid, int nthreads) {
  for (int c = tid; c < 512; c += nthreads) {
    int row = c >> 2, part = c & 3;
    async_cp16(src_base + (size_t)row * ld + (size_t)k0 + part * 8,
               lds + row * LDS_STRIDE + part * 8);
  }
}

// ---------------- utility kernels ----------------
__global__ void zero_f32_kernel(float* p, size_t n) {
  size_t i = (size_t)blockIdx.x * blockDim.x + threadIdx.x;
  size_t stride = (size_t)gridDim.x * blockDim.x;
  for (; i < n; i += stride) p[i] = 0.0f;
}

__global__ void zero_u32_kernel(unsigned* p, int n) {
  int i = blockIdx.x * blockDim.x + threadIdx.x;
  if (i < n) p[i] = 0u;
}

__global__ void cvt_bf16_kernel(const float* src, __bf16* dst, size_t n) {
  size_t i = (size_t)blockIdx.x * blockDim.x + threadIdx.x;
  size_t stride = (size_t)gridDim.x * blockDim.x;
  for (; i < n; i += stride) dst[i] = (__bf16)src[i];
}

// ---------------- RMSNorm (f32 in -> bf16 out) ----------------
__global__ __launch_bounds__(256) void rmsnorm_bf16_kernel(const float* x, const float* w,
                                                           __bf16* out, int D_) {
  int row = blockIdx.x;
  const float* xr = x + (size_t)row * D_;
  float s = 0.0f;
  for (int i = threadIdx.x; i < D_; i += 256) {
    float v = xr[i];
    s += v * v;
  }
  for (int d = 16; d >= 1; d >>= 1) s += __shfl_xor(s, d, 32);
  __shared__ float red[8];
  if ((threadIdx.x & 31) == 0) red[threadIdx.x >> 5] = s;
  __syncthreads();
  float tot = 0.0f;
  for (int i = 0; i < 8; ++i) tot += red[i];
  float r = rsqrtf(tot / (float)D_ + 1e-5f);
  for (int i = threadIdx.x; i < D_; i += 256)
    out[(size_t)row * D_ + i] = (__bf16)(xr[i] * r * w[i]);
}

// ---------------- generic bf16 GEMM: C[M,N] = A[M,K] * B[N,K]^T (+addend) ----------------
// 128 threads = 4 waves (2x2), 128x128 block tile, 64x64 wave tile, 16 WMMA / k-step / wave
__global__ __launch_bounds__(128) void gemm_bf16_kernel(const __bf16* A, const __bf16* Bm,
                                                        const float* addend, float* C,
                                                        int M, int N, int Kd) {
  __shared__ __bf16 sA[128 * LDS_STRIDE];
  __shared__ __bf16 sB[128 * LDS_STRIDE];
  int tid = threadIdx.x, lane = tid & 31, wave = tid >> 5;
  int wm = wave & 1, wn = wave >> 1;  // 2 x 2 wave grid
  int m0 = blockIdx.y * 128, n0 = blockIdx.x * 128;

  v8f acc[4][4];
  for (int i = 0; i < 4; ++i)
    for (int j = 0; j < 4; ++j) acc[i][j] = v8f_zero();

  for (int k0 = 0; k0 < Kd; k0 += 32) {
    __syncthreads();
    stage_rows_async(A + (size_t)m0 * Kd, Kd, k0, sA, tid, 128);
    stage_rows_async(Bm + (size_t)n0 * Kd, Kd, k0, sB, tid, 128);
    async_wait0();
    __syncthreads();
    v16bf af[4], bfr[4];
    for (int i = 0; i < 4; ++i) af[i] = frag_a(sA, wm * 64 + i * 16, lane);
    for (int j = 0; j < 4; ++j) bfr[j] = frag_b(sB, wn * 64 + j * 16, lane);
    for (int i = 0; i < 4; ++i)
      for (int j = 0; j < 4; ++j) acc[i][j] = wmma_bf16(af[i], bfr[j], acc[i][j]);
  }

  int mg = m0 + wm * 64, ng = n0 + wn * 64;
  for (int i = 0; i < 4; ++i)
    for (int j = 0; j < 4; ++j)
      for (int r = 0; r < 8; ++r) {
        int row = mg + i * 16 + r + 8 * (lane >> 4);
        int col = ng + j * 16 + (lane & 15);
        float v = acc[i][j][r];
        if (addend) v += addend[(size_t)row * N + col];
        C[(size_t)row * N + col] = v;
      }
}

// ---------------- RoPE + layout to [b,h,s,hd] bf16 ----------------
__global__ void rope_kernel(const float* qkv, const float* cosb, const float* sinb,
                            __bf16* qo, __bf16* ko, __bf16* vo) {
  size_t idx = (size_t)blockIdx.x * blockDim.x + threadIdx.x;
  if (idx >= (size_t)Tc * 32 * 128) return;
  int i  = (int)(idx & 127);
  int hh = (int)((idx >> 7) & 31);
  int t  = (int)(idx >> 12);
  int b  = t >> 11;        // S = 2048
  int s_ = t & 2047;
  const float* base = qkv + (size_t)t * QKVF + hh * 128;
  float v = base[i];
  if (hh < NHc + NKVc) {  // rope for q and k
    float c  = cosb[s_ * 128 + i];
    float sn = sinb[s_ * 128 + i];
    float other = (i < 64) ? -base[i + 64] : base[i - 64];
    float r = v * c + other * sn;
    if (hh < NHc) {
      qo[(((size_t)b * NHc + hh) * Sc + s_) * 128 + i] = (__bf16)r;
    } else {
      int kh = hh - NHc;
      ko[(((size_t)b * NKVc + kh) * Sc + s_) * 128 + i] = (__bf16)r;
    }
  } else {
    int vh = hh - NHc - NKVc;
    vo[(((size_t)b * NKVc + vh) * Sc + s_) * 128 + i] = (__bf16)v;
  }
}

// ---------------- flash attention (causal, GQA rep=2) ----------------
__global__ __launch_bounds__(128) void attn_kernel(const __bf16* Q, const __bf16* Kb_,
                                                   const __bf16* V, __bf16* Oout) {
  const int qt = blockIdx.x, h = blockIdx.y, b = blockIdx.z;
  const int kvh = h >> 1;
  const int tid = threadIdx.x, lane = tid & 31, wave = tid >> 5;
  const int q0 = qt * 64;
  const int qrow0 = q0 + wave * 16;
  const __bf16* Qb = Q   + ((size_t)b * NHc  + h)   * Sc * HDc;
  const __bf16* Kb = Kb_ + ((size_t)b * NKVc + kvh) * Sc * HDc;
  const __bf16* Vb = V   + ((size_t)b * NKVc + kvh) * Sc * HDc;

  __shared__ __bf16 sVt[128][72];     // V^T tile: [hd][key], 64 keys
  __shared__ __bf16 sP[4][16][72];    // per-wave P tile [m][key]

  // Q fragments, direct from global (hd-contiguous)
  v16bf qf[4];
  {
    int m = lane & 15, kk = (lane >> 4) * 8;
    for (int c = 0; c < 4; ++c) {
      const __bf16* p = Qb + (size_t)(qrow0 + m) * HDc + c * 32 + kk;
      v16bf a;
      ((uint4*)&a)[0] = *(const uint4*)(p);
      ((uint4*)&a)[1] = *(const uint4*)(p + 16);
      qf[c] = a;
    }
  }

  float mstat[8], lstat[8];
  v8f oacc[8];
  for (int r = 0; r < 8; ++r) { mstat[r] = -1e30f; lstat[r] = 0.0f; }
  for (int j = 0; j < 8; ++j) oacc[j] = v8f_zero();

  const int kmax = q0 + 63;
  const float scale = 0.08838834764831845f;  // 1/sqrt(128)

  for (int kb = 0; kb <= kmax; kb += 64) {
    __syncthreads();
    // stage transposed V tile
    for (int c = tid; c < 1024; c += 128) {
      int krow = c >> 4, part = c & 15;
      const __bf16* src = Vb + (size_t)(kb + krow) * HDc + part * 8;
      __bf16 tmp[8];
      *(uint4*)tmp = *(const uint4*)src;
      for (int j = 0; j < 8; ++j) sVt[part * 8 + j][krow] = tmp[j];
    }
    __syncthreads();

    // S = Q * K^T (B fragments direct from global: key rows are hd-contiguous)
    v8f sacc[4];
    for (int nst = 0; nst < 4; ++nst) {
      v8f a = v8f_zero();
      int n = kb + nst * 16 + (lane & 15);
      int kk = (lane >> 4) * 8;
      for (int kc = 0; kc < 4; ++kc) {
        const __bf16* p = Kb + (size_t)n * HDc + kc * 32 + kk;
        v16bf bfr;
        ((uint4*)&bfr)[0] = *(const uint4*)(p);
        ((uint4*)&bfr)[1] = *(const uint4*)(p + 16);
        a = wmma_bf16(qf[kc], bfr, a);
      }
      sacc[nst] = a;
    }

    // online softmax
    for (int r = 0; r < 8; ++r) {
      int qi = qrow0 + r + 8 * (lane >> 4);
      float mx = -1e30f;
      for (int nst = 0; nst < 4; ++nst) {
        int key = kb + nst * 16 + (lane & 15);
        float s = sacc[nst][r] * scale;
        if (key > qi) s = -1e30f;
        sacc[nst][r] = s;
        mx = fmaxf(mx, s);
      }
      for (int d = 1; d < 16; d <<= 1) mx = fmaxf(mx, __shfl_xor(mx, d, 32));
      float newm = fmaxf(mstat[r], mx);
      float rsum = 0.0f;
      for (int nst = 0; nst < 4; ++nst) {
        float p = __expf(sacc[nst][r] - newm);
        sacc[nst][r] = p;
        rsum += p;
      }
      for (int d = 1; d < 16; d <<= 1) rsum += __shfl_xor(rsum, d, 32);
      float alpha = __expf(mstat[r] - newm);
      lstat[r] = lstat[r] * alpha + rsum;
      mstat[r] = newm;
      for (int j = 0; j < 8; ++j) oacc[j][r] *= alpha;
      int mrow = r + 8 * (lane >> 4);
      for (int nst = 0; nst < 4; ++nst)
        sP[wave][mrow][nst * 16 + (lane & 15)] = (__bf16)sacc[nst][r];
    }
    __syncthreads();  // P visible (and Vt still live)

    // O += P * V
    v16bf pf[2];
    {
      int m = lane & 15, kk = (lane >> 4) * 8;
      for (int c = 0; c < 2; ++c) {
        v16bf a;
        ((uint4*)&a)[0] = *(const uint4*)(&sP[wave][m][c * 32 + kk]);
        ((uint4*)&a)[1] = *(const uint4*)(&sP[wave][m][c * 32 + kk + 16]);
        pf[c] = a;
      }
    }
    for (int j = 0; j < 8; ++j) {
      int n = j * 16 + (lane & 15);
      int kk = (lane >> 4) * 8;
      for (int kc = 0; kc < 2; ++kc) {
        v16bf bfr;
        ((uint4*)&bfr)[0] = *(const uint4*)(&sVt[n][kc * 32 + kk]);
        ((uint4*)&bfr)[1] = *(const uint4*)(&sVt[n][kc * 32 + kk + 16]);
        oacc[j] = wmma_bf16(pf[kc], bfr, oacc[j]);
      }
    }
  }

  // epilogue: normalize and store [t, h*HD + hd] bf16
  for (int j = 0; j < 8; ++j)
    for (int r = 0; r < 8; ++r) {
      int qi = qrow0 + r + 8 * (lane >> 4);
      size_t t = (size_t)b * Sc + qi;
      float v = oacc[j][r] / lstat[r];
      Oout[t * (NHc * HDc) + h * HDc + j * 16 + (lane & 15)] = (__bf16)v;
    }
}

// ---------------- MoE gate: softmax + top-2 ----------------
__global__ __launch_bounds__(256) void gate_kernel(const __bf16* x, const float* gw,
                                                   int* topi, float* topw, unsigned* cnt) {
  int wave = threadIdx.x >> 5, lane = threadIdx.x & 31;
  int t = blockIdx.x * 8 + wave;
  float acc[Ec];
  for (int e = 0; e < Ec; ++e) acc[e] = 0.0f;
  for (int d = lane; d < Dc; d += 32) {
    float xv = (float)x[(size_t)t * Dc + d];
    for (int e = 0; e < Ec; ++e) acc[e] += xv * gw[e * Dc + d];
  }
  for (int e = 0; e < Ec; ++e)
    for (int d = 16; d >= 1; d >>= 1) acc[e] += __shfl_xor(acc[e], d, 32);
  if (lane == 0) {
    float mx = acc[0];
    for (int e = 1; e < Ec; ++e) mx = fmaxf(mx, acc[e]);
    float p[Ec], sum = 0.0f;
    for (int e = 0; e < Ec; ++e) { p[e] = __expf(acc[e] - mx); sum += p[e]; }
    for (int e = 0; e < Ec; ++e) p[e] /= sum;
    int e0 = 0; float p0 = p[0];
    for (int e = 1; e < Ec; ++e) if (p[e] > p0) { p0 = p[e]; e0 = e; }
    int e1 = (e0 == 0) ? 1 : 0; float p1 = p[e1];
    for (int e = 0; e < Ec; ++e) if (e != e0 && p[e] > p1) { p1 = p[e]; e1 = e; }
    float ws = p0 + p1;
    topi[t * 2 + 0] = e0; topw[t * 2 + 0] = p0 / ws;
    topi[t * 2 + 1] = e1; topw[t * 2 + 1] = p1 / ws;
    atomicAdd(&cnt[e0], 1u);
    atomicAdd(&cnt[e1], 1u);
  }
}

__global__ void offsets_kernel(const unsigned* cnt, unsigned* off, unsigned* cursor) {
  if (threadIdx.x == 0 && blockIdx.x == 0) {
    unsigned a = 0;
    for (int e = 0; e < Ec; ++e) { off[e] = a; a += cnt[e]; cursor[e] = 0u; }
  }
}

__global__ void scatter_kernel(const int* topi, const float* topw, const unsigned* off,
                               unsigned* cursor, int* tokids, float* tokw) {
  int idx = blockIdx.x * blockDim.x + threadIdx.x;
  if (idx >= Tc * Kc) return;
  int e = topi[idx];
  unsigned pos = atomicAdd(&cursor[e], 1u);
  tokids[off[e] + pos] = idx >> 1;
  tokw[off[e] + pos] = topw[idx];
}

// ---------------- MoE GEMM 1: GU = silu(X W1^T) * (X W3^T), gathered rows ----------------
__global__ __launch_bounds__(256) void moe_gemm13_kernel(const __bf16* X, const __bf16* W1,
                                                         const __bf16* W3, const int* tokids,
                                                         const unsigned* cnt, const unsigned* off,
                                                         __bf16* GU) {
  int e = blockIdx.z;
  int count = (int)cnt[e];
  int base = (int)off[e];
  int m0 = blockIdx.y * 128;
  if (m0 >= count) return;
  int n0 = blockIdx.x * 128;
  const __bf16* B1 = W1 + (size_t)e * Fc * Dc;
  const __bf16* B3 = W3 + (size_t)e * Fc * Dc;

  __shared__ __bf16 sA[128 * LDS_STRIDE];
  __shared__ __bf16 sB1[128 * LDS_STRIDE];
  __shared__ __bf16 sB3[128 * LDS_STRIDE];

  int tid = threadIdx.x, lane = tid & 31, wave = tid >> 5;
  int wm = wave & 1, wn = wave >> 1;  // 2 x 4

  v8f a1[4][2], a3[4][2];
  for (int i = 0; i < 4; ++i)
    for (int j = 0; j < 2; ++j) { a1[i][j] = v8f_zero(); a3[i][j] = v8f_zero(); }

  for (int k0 = 0; k0 < Dc; k0 += 32) {
    __syncthreads();
    // gathered A stage (async)
    for (int c = tid; c < 512; c += 256) {
      int row = c >> 2, part = c & 3;
      int rl = m0 + row;
      if (rl >= count) rl = count - 1;
      int tok = tokids[base + rl];
      async_cp16(X + (size_t)tok * Dc + k0 + part * 8,
                 sA + row * LDS_STRIDE + part * 8);
    }
    stage_rows_async(B1 + (size_t)n0 * Dc, Dc, k0, sB1, tid, 256);
    stage_rows_async(B3 + (size_t)n0 * Dc, Dc, k0, sB3, tid, 256);
    async_wait0();
    __syncthreads();
    v16bf af[4], b1f[2], b3f[2];
    for (int i = 0; i < 4; ++i) af[i] = frag_a(sA, wm * 64 + i * 16, lane);
    for (int j = 0; j < 2; ++j) {
      b1f[j] = frag_b(sB1, wn * 32 + j * 16, lane);
      b3f[j] = frag_b(sB3, wn * 32 + j * 16, lane);
    }
    for (int i = 0; i < 4; ++i)
      for (int j = 0; j < 2; ++j) {
        a1[i][j] = wmma_bf16(af[i], b1f[j], a1[i][j]);
        a3[i][j] = wmma_bf16(af[i], b3f[j], a3[i][j]);
      }
  }

  int mg = m0 + wm * 64, ng = n0 + wn * 32;
  for (int i = 0; i < 4; ++i)
    for (int j = 0; j < 2; ++j)
      for (int r = 0; r < 8; ++r) {
        int row = mg + i * 16 + r + 8 * (lane >> 4);
        if (row >= count) continue;
        int col = ng + j * 16 + (lane & 15);
        float g = a1[i][j][r];
        g = g / (1.0f + __expf(-g));
        GU[(size_t)(base + row) * Fc + col] = (__bf16)(g * a3[i][j][r]);
      }
}

// ---------------- MoE GEMM 2: out[tok] += w * (GU W2^T) ----------------
__global__ __launch_bounds__(256) void moe_gemm2_kernel(const __bf16* GU, const __bf16* W2,
                                                        const int* tokids, const float* tokw,
                                                        const unsigned* cnt, const unsigned* off,
                                                        float* out) {
  int e = blockIdx.z;
  int count = (int)cnt[e];
  int base = (int)off[e];
  int m0 = blockIdx.y * 128;
  if (m0 >= count) return;
  int n0 = blockIdx.x * 128;
  const __bf16* Bm = W2 + (size_t)e * Dc * Fc;

  __shared__ __bf16 sA[128 * LDS_STRIDE];
  __shared__ __bf16 sB[128 * LDS_STRIDE];

  int tid = threadIdx.x, lane = tid & 31, wave = tid >> 5;
  int wm = wave & 1, wn = wave >> 1;

  v8f acc[4][2];
  for (int i = 0; i < 4; ++i)
    for (int j = 0; j < 2; ++j) acc[i][j] = v8f_zero();

  for (int k0 = 0; k0 < Fc; k0 += 32) {
    __syncthreads();
    for (int c = tid; c < 512; c += 256) {
      int row = c >> 2, part = c & 3;
      int rl = m0 + row;
      if (rl >= count) rl = count - 1;
      async_cp16(GU + (size_t)(base + rl) * Fc + k0 + part * 8,
                 sA + row * LDS_STRIDE + part * 8);
    }
    stage_rows_async(Bm + (size_t)n0 * Fc, Fc, k0, sB, tid, 256);
    async_wait0();
    __syncthreads();
    v16bf af[4], bfr[2];
    for (int i = 0; i < 4; ++i) af[i] = frag_a(sA, wm * 64 + i * 16, lane);
    for (int j = 0; j < 2; ++j) bfr[j] = frag_b(sB, wn * 32 + j * 16, lane);
    for (int i = 0; i < 4; ++i)
      for (int j = 0; j < 2; ++j) acc[i][j] = wmma_bf16(af[i], bfr[j], acc[i][j]);
  }

  int mg = m0 + wm * 64, ng = n0 + wn * 32;
  for (int i = 0; i < 4; ++i)
    for (int j = 0; j < 2; ++j)
      for (int r = 0; r < 8; ++r) {
        int row = mg + i * 16 + r + 8 * (lane >> 4);
        if (row >= count) continue;
        int col = ng + j * 16 + (lane & 15);
        int tok = tokids[base + row];
        float w = tokw[base + row];
        atomicAdd(&out[(size_t)tok * Dc + col], w * acc[i][j][r]);
      }
}

// ---------------- host launcher ----------------
extern "C" void kernel_launch(void* const* d_in, const int* in_sizes, int n_in,
                              void* d_out, int out_size, void* d_ws, size_t ws_size,
                              hipStream_t stream) {
  (void)in_sizes; (void)n_in; (void)out_size; (void)ws_size;
  const float* hidden = (const float*)d_in[0];
  const float* cosb   = (const float*)d_in[1];
  const float* sinb   = (const float*)d_in[2];
  const float* ln1w   = (const float*)d_in[3];
  const float* ln2w   = (const float*)d_in[4];
  const float* wqkv   = (const float*)d_in[5];
  const float* wo     = (const float*)d_in[6];
  const float* gatew  = (const float*)d_in[7];
  const float* w1     = (const float*)d_in[8];
  const float* w3     = (const float*)d_in[9];
  const float* w2     = (const float*)d_in[10];

  float* out   = (float*)d_out;                    // [T, D]
  float* resid = out + (size_t)Tc * Dc;            // [T, D]

  char* ws = (char*)d_ws;
  auto alloc = [&](size_t bytes) {
    void* p = (void*)ws;
    ws += (bytes + 255) & ~(size_t)255;
    return p;
  };

  __bf16* xb     = (__bf16*)alloc((size_t)Tc * Dc * 2);
  __bf16* wqkvb  = (__bf16*)alloc((size_t)QKVF * Dc * 2);
  float*  qkvf   = (float*) alloc((size_t)Tc * QKVF * 4);
  __bf16* qb     = (__bf16*)alloc((size_t)Bc * NHc * Sc * HDc * 2);
  __bf16* kbuf   = (__bf16*)alloc((size_t)Bc * NKVc * Sc * HDc * 2);
  __bf16* vbuf   = (__bf16*)alloc((size_t)Bc * NKVc * Sc * HDc * 2);
  __bf16* attnb  = (__bf16*)alloc((size_t)Tc * NHc * HDc * 2);
  __bf16* wob    = (__bf16*)alloc((size_t)Dc * NHc * HDc * 2);
  __bf16* x2b    = (__bf16*)alloc((size_t)Tc * Dc * 2);
  int*     topi  = (int*)    alloc((size_t)Tc * Kc * 4);
  float*   topw  = (float*)  alloc((size_t)Tc * Kc * 4);
  unsigned* cnt    = (unsigned*)alloc(Ec * 4);
  unsigned* off    = (unsigned*)alloc(Ec * 4);
  unsigned* cursor = (unsigned*)alloc(Ec * 4);
  int*     tokids = (int*)   alloc((size_t)Tc * Kc * 4);
  float*   tokw   = (float*) alloc((size_t)Tc * Kc * 4);
  __bf16* w1b = (__bf16*)alloc((size_t)Ec * Fc * Dc * 2);
  __bf16* w3b = (__bf16*)alloc((size_t)Ec * Fc * Dc * 2);
  __bf16* w2b = (__bf16*)alloc((size_t)Ec * Dc * Fc * 2);
  __bf16* gub = (__bf16*)alloc((size_t)Tc * Kc * Fc * 2);

  // zero MoE output and routing counts
  zero_f32_kernel<<<4096, 256, 0, stream>>>(out, (size_t)Tc * Dc);
  zero_u32_kernel<<<1, 32, 0, stream>>>(cnt, Ec);

  // weight conversion fp32 -> bf16
  cvt_bf16_kernel<<<4096, 256, 0, stream>>>(wqkv, wqkvb, (size_t)QKVF * Dc);
  cvt_bf16_kernel<<<4096, 256, 0, stream>>>(wo, wob, (size_t)Dc * NHc * HDc);
  cvt_bf16_kernel<<<8192, 256, 0, stream>>>(w1, w1b, (size_t)Ec * Fc * Dc);
  cvt_bf16_kernel<<<8192, 256, 0, stream>>>(w3, w3b, (size_t)Ec * Fc * Dc);
  cvt_bf16_kernel<<<8192, 256, 0, stream>>>(w2, w2b, (size_t)Ec * Dc * Fc);

  // ln1 + QKV projection
  rmsnorm_bf16_kernel<<<Tc, 256, 0, stream>>>(hidden, ln1w, xb, Dc);
  gemm_bf16_kernel<<<dim3(QKVF / 128, Tc / 128), 128, 0, stream>>>(
      xb, wqkvb, nullptr, qkvf, Tc, QKVF, Dc);

  // RoPE + relayout
  rope_kernel<<<(Tc * 32 * 128) / 256, 256, 0, stream>>>(qkvf, cosb, sinb, qb, kbuf, vbuf);

  // flash attention
  attn_kernel<<<dim3(Sc / 64, NHc, Bc), 128, 0, stream>>>(qb, kbuf, vbuf, attnb);

  // output projection + residual (written to second output)
  gemm_bf16_kernel<<<dim3(Dc / 128, Tc / 128), 128, 0, stream>>>(
      attnb, wob, hidden, resid, Tc, Dc, NHc * HDc);

  // ln2 + MoE routing
  rmsnorm_bf16_kernel<<<Tc, 256, 0, stream>>>(resid, ln2w, x2b, Dc);
  gate_kernel<<<Tc / 8, 256, 0, stream>>>(x2b, gatew, topi, topw, cnt);
  offsets_kernel<<<1, 1, 0, stream>>>(cnt, off, cursor);
  scatter_kernel<<<(Tc * Kc + 255) / 256, 256, 0, stream>>>(topi, topw, off, cursor,
                                                            tokids, tokw);

  // expert GEMMs (gathered tokens only)
  moe_gemm13_kernel<<<dim3(Fc / 128, Tc / 128, Ec), 256, 0, stream>>>(
      x2b, w1b, w3b, tokids, cnt, off, gub);
  moe_gemm2_kernel<<<dim3(Dc / 128, Tc / 128, Ec), 256, 0, stream>>>(
      gub, w2b, tokids, tokw, cnt, off, out);
}
